// AMIPRouter_33767032881177
// MI455X (gfx1250) — compile-verified, matrix operand order checked by reading, not run on hardware
//
#include <hip/hip_runtime.h>
#include <hip/hip_bf16.h>

// ---------------------------------------------------------------------------
// Problem constants (match reference): B=2, S=2048, D=2048, K=8, F=D/4=512, R=8
// ---------------------------------------------------------------------------
constexpr int Bc = 2;
constexpr int Sc = 2048;
constexpr int Dc = 2048;
constexpr int Kc = 8;
constexpr int Fc = Dc / 4;   // 512
constexpr int Rc = 8;

constexpr int KC   = 64;     // k-elements staged per barrier (2 WMMA k-steps)
constexpr int MT   = 64;     // macro-tile M
constexpr int NT   = 128;    // macro-tile N
constexpr int LDST = 72;     // padded LDS row stride (bf16): 144B = 36 banks,
                             // gcd(36,64)=4 -> 16 fragment rows in 16 distinct bank groups

typedef __attribute__((ext_vector_type(16))) __bf16 v16bf;
typedef __attribute__((ext_vector_type(8)))  __bf16 v8bf;
typedef __attribute__((ext_vector_type(8)))  float  v8f;

// ---------------------------------------------------------------------------
// Workgroup barrier that orders ONLY LDS traffic (split-counter wait), leaving
// global loads in flight across the barrier so they retire under the WMMAs.
// (__syncthreads() would emit s_wait_loadcnt_dscnt 0 and drain the prefetched
// next-stage tile before compute.)
// ---------------------------------------------------------------------------
__device__ inline void wg_barrier_lds() {
    asm volatile("s_wait_dscnt 0x0" ::: "memory");
    __builtin_amdgcn_s_barrier();
}

// ---------------------------------------------------------------------------
// Fragment gather from LDS matching the CDNA5 16-bit WMMA operand layout:
// lanes 0-15  hold K = {ks+0..7, ks+16..23} (two contiguous 16B chunks)
// lanes 16-31 hold K = {ks+8..15, ks+24..31}
// ---------------------------------------------------------------------------
__device__ inline v16bf load_frag(const __bf16* lds, int row, int ks, int lane) {
    const __bf16* p = lds + row * LDST + ks + ((lane & 16) ? 8 : 0);
    v16bf f;
#pragma unroll
    for (int i = 0; i < 8; ++i) { f[i] = p[i]; f[i + 8] = p[i + 16]; }
    return f;
}

// ---------------------------------------------------------------------------
// Pre-pass A: elementwise f32 -> bf16 (h_L)
// ---------------------------------------------------------------------------
__global__ __launch_bounds__(256) void cvt_kernel(
    const float* __restrict__ src, __bf16* __restrict__ dst)
{
    size_t i = ((size_t)blockIdx.x * 256 + threadIdx.x) * 8;
    v8bf t;
#pragma unroll
    for (int j = 0; j < 8; ++j) t[j] = (__bf16)src[i + j];
    *(v8bf*)&dst[i] = t;
}

// ---------------------------------------------------------------------------
// Pre-pass B: tiled transpose + f32->bf16.  src: [z][Rr][Cc] f32 row-major,
// dst: [z][Cc][Rr] bf16.  64x64 tile per WG, coalesced both sides via LDS.
// ---------------------------------------------------------------------------
__global__ __launch_bounds__(256) void transpose_cvt_kernel(
    const float* __restrict__ src, __bf16* __restrict__ dst, int Rr, int Cc)
{
    __shared__ __bf16 t[64][65];
    const float* s = src + (size_t)blockIdx.z * Rr * Cc;
    __bf16*      d = dst + (size_t)blockIdx.z * Rr * Cc;
    int r0 = blockIdx.y * 64, c0 = blockIdx.x * 64;
    int tid = threadIdx.x;
    int lr = tid >> 2, lc = (tid & 3) * 16;
    const float* p = s + (size_t)(r0 + lr) * Cc + c0 + lc;
#pragma unroll
    for (int j = 0; j < 16; ++j) t[lc + j][lr] = (__bf16)p[j];
    __syncthreads();
    __bf16* q = d + (size_t)(c0 + lr) * Rr + r0 + lc;
#pragma unroll
    for (int j = 0; j < 16; ++j) q[j] = t[lr][lc + j];
}

// ---------------------------------------------------------------------------
// Kernel 1: routing softmax w[b,s,k] + windowed unmasked-neighbor count cnt[b,s]
// One wave per token; 8 waves per block. masked: jax bool -> 1 byte.
// ---------------------------------------------------------------------------
__global__ __launch_bounds__(256) void route_kernel(
    const float* __restrict__ h, const unsigned char* __restrict__ masked,
    const float* __restrict__ Wr, const float* __restrict__ br,
    float* __restrict__ w, float* __restrict__ cnt)
{
    int tid = threadIdx.x, lane = tid & 31, wv = tid >> 5;
    int token = blockIdx.x * 8 + wv;                 // 0 .. B*S-1
    const float* hp = h + (size_t)token * Dc;

    float acc[Kc];
#pragma unroll
    for (int e = 0; e < Kc; ++e) acc[e] = 0.f;

    for (int d = lane; d < Dc; d += 32) {
        float hv = hp[d];
        const float* wr = Wr + (size_t)d * Kc;
#pragma unroll
        for (int e = 0; e < Kc; ++e) acc[e] += hv * wr[e];
    }
#pragma unroll
    for (int off = 16; off; off >>= 1) {
#pragma unroll
        for (int e = 0; e < Kc; ++e) acc[e] += __shfl_xor(acc[e], off, 32);
    }
    if (lane == 0) {
        float m = -3.4e38f;
#pragma unroll
        for (int e = 0; e < Kc; ++e) { acc[e] += br[e]; m = fmaxf(m, acc[e]); }
        float sum = 0.f;
#pragma unroll
        for (int e = 0; e < Kc; ++e) { acc[e] = __expf(acc[e] - m); sum += acc[e]; }
        float inv = 1.f / sum;
#pragma unroll
        for (int e = 0; e < Kc; ++e) w[(size_t)token * Kc + e] = acc[e] * inv;

        int b = token / Sc, s = token % Sc;
        int lo = max(s - Rc, 0), hi = min(s + Rc, Sc - 1);
        float c = 0.f;
        for (int s2 = lo; s2 <= hi; ++s2) c += masked[b * Sc + s2] ? 0.f : 1.f;
        c -= masked[token] ? 0.f : 1.f;              // exclude self
        cnt[token] = c;
    }
}

// ---------------------------------------------------------------------------
// GEMM core (shared shape for kernels 2 & 3): 64(M) x 128(N) x k-step 64,
// 256 threads = 8 waves in 2(m) x 4(n); each wave owns a 32x32 block (2x2
// 16x16 subtiles, 4 f32 accumulators). Double-buffered LDS: global loads for
// stage i+1 are issued BEFORE the LDS-only barrier + WMMA block of stage i,
// so HBM latency hides under matrix math; the s_wait_loadcnt lands only at
// the DS-store commit after compute (register dependency).
// ---------------------------------------------------------------------------
#define GEMM_BODY(LDK, KDIM)                                                        \
    const __bf16* Asrc = Ag + (size_t)arow * (LDK) + acb;                           \
    const __bf16* Bsrc = Bg + (size_t)brow * (LDK) + bcb;                           \
    v8bf ra0 = *(const v8bf*)(Asrc);                                                \
    v8bf ra1 = *(const v8bf*)(Asrc + 8);                                            \
    v8bf rb0 = *(const v8bf*)(Bsrc);                                                \
    v8bf rb1 = *(const v8bf*)(Bsrc + 8);                                            \
    v8bf rb2 = *(const v8bf*)(Bsrc + 16);                                           \
    v8bf rb3 = *(const v8bf*)(Bsrc + 24);                                           \
    {                                                                               \
        __bf16* dA = &sA[0][arow * LDST + acb];                                     \
        *(v8bf*)dA = ra0; *(v8bf*)(dA + 8) = ra1;                                   \
        __bf16* dB = &sB[0][brow * LDST + bcb];                                     \
        *(v8bf*)dB = rb0; *(v8bf*)(dB + 8) = rb1;                                   \
        *(v8bf*)(dB + 16) = rb2; *(v8bf*)(dB + 24) = rb3;                           \
    }                                                                               \
    int cur = 0;                                                                    \
    for (int kk = KC;; kk += KC) {                                                  \
        bool more = kk < (KDIM);                                                    \
        if (more) {                                                                 \
            ra0 = *(const v8bf*)(Asrc + kk);                                        \
            ra1 = *(const v8bf*)(Asrc + kk + 8);                                    \
            rb0 = *(const v8bf*)(Bsrc + kk);                                        \
            rb1 = *(const v8bf*)(Bsrc + kk + 8);                                    \
            rb2 = *(const v8bf*)(Bsrc + kk + 16);                                   \
            rb3 = *(const v8bf*)(Bsrc + kk + 24);                                   \
            if (kk + KC < (KDIM)) {                                                 \
                __builtin_prefetch(Asrc + kk + KC, 0, 1);                           \
                __builtin_prefetch(Bsrc + kk + KC, 0, 1);                           \
            }                                                                       \
        }                                                                           \
        wg_barrier_lds();                                                           \
        const __bf16* pA = sA[cur];                                                 \
        const __bf16* pB = sB[cur];                                                 \
        int r = lane & 15;                                                          \
        _Pragma("unroll")                                                           \
        for (int ks = 0; ks < KC; ks += 32) {                                       \
            v16bf a0 = load_frag(pA, m0 + r,      ks, lane);                        \
            v16bf a1 = load_frag(pA, m0 + 16 + r, ks, lane);                        \
            v16bf bb0 = load_frag(pB, n0 + r,      ks, lane);                       \
            v16bf bb1 = load_frag(pB, n0 + 16 + r, ks, lane);                       \
            c00 = __builtin_amdgcn_wmma_f32_16x16x32_bf16(false, a0, false, bb0, (short)0, c00, false, false); \
            c01 = __builtin_amdgcn_wmma_f32_16x16x32_bf16(false, a0, false, bb1, (short)0, c01, false, false); \
            c10 = __builtin_amdgcn_wmma_f32_16x16x32_bf16(false, a1, false, bb0, (short)0, c10, false, false); \
            c11 = __builtin_amdgcn_wmma_f32_16x16x32_bf16(false, a1, false, bb1, (short)0, c11, false, false); \
        }                                                                           \
        if (!more) break;                                                           \
        int nxt = cur ^ 1;                                                          \
        {                                                                           \
            __bf16* dA = &sA[nxt][arow * LDST + acb];                               \
            *(v8bf*)dA = ra0; *(v8bf*)(dA + 8) = ra1;                               \
            __bf16* dB = &sB[nxt][brow * LDST + bcb];                               \
            *(v8bf*)dB = rb0; *(v8bf*)(dB + 8) = rb1;                               \
            *(v8bf*)(dB + 16) = rb2; *(v8bf*)(dB + 24) = rb3;                       \
        }                                                                           \
        cur = nxt;                                                                  \
    }

// Kernel 2: hid[b,k,s,f] = gelu_erf( h[b,s,:] @ W1[k,:,:] + b1[k,:] )
//   A = hA[b] (bf16 [S][D]); B = W1T[k] (bf16 [F][D], k-dim contiguous)
__global__ __launch_bounds__(256) void gemm1_kernel(
    const __bf16* __restrict__ hA, const __bf16* __restrict__ W1T,
    const float* __restrict__ b1, __bf16* __restrict__ hid)
{
    __shared__ __align__(16) __bf16 sA[2][MT * LDST];
    __shared__ __align__(16) __bf16 sB[2][NT * LDST];

    int z = blockIdx.z;              // b*K + k
    int k = z & (Kc - 1);
    int b = z >> 3;
    int s0 = blockIdx.x * MT;
    int f0 = blockIdx.y * NT;

    const __bf16* Ag = hA  + (size_t)(b * Sc + s0) * Dc;
    const __bf16* Bg = W1T + (size_t)k * Fc * Dc + (size_t)f0 * Dc;

    int tid = threadIdx.x, lane = tid & 31, wv = tid >> 5;
    int m0 = (wv & 1) * 32;
    int n0 = (wv >> 1) * 32;
    v8f c00 = {}, c01 = {}, c10 = {}, c11 = {};

    const int arow = tid >> 2, acb = (tid & 3) * 16;   // A: 16 bf16 per thread
    const int brow = tid >> 1, bcb = (tid & 1) * 32;   // B: 32 bf16 per thread

    GEMM_BODY(Dc, Dc)

    // epilogue: +bias, exact-erf GELU, store bf16
    int col = lane & 15;
    int rbase = (lane & 16) ? 8 : 0;
    __bf16* hp = hid + (size_t)z * Sc * Fc;
    v8f* cs[4] = { &c00, &c01, &c10, &c11 };
#pragma unroll
    for (int t = 0; t < 4; ++t) {
        int mrow = s0 + m0 + (t >> 1) * 16 + rbase;
        int fcol = f0 + n0 + (t & 1) * 16 + col;
        float bias = b1[k * Fc + fcol];
#pragma unroll
        for (int i = 0; i < 8; ++i) {
            float x = (*cs[t])[i] + bias;
            float g = 0.5f * x * (1.f + erff(x * 0.70710678118654752f));
            hp[(size_t)(mrow + i) * Fc + fcol] = (__bf16)g;
        }
    }
}

// Kernel 3: contrib[b,s,k,d] = unm[b,s] * ( hid[b,k,s,:] @ W2[k,:,:] + b2[k,:] )
//   A = hid[z] (bf16 [S][F]); B = W2T[k] (bf16 [D][F], k-dim contiguous)
__global__ __launch_bounds__(256) void gemm2_kernel(
    const __bf16* __restrict__ hid, const __bf16* __restrict__ W2T,
    const float* __restrict__ b2, const unsigned char* __restrict__ masked,
    float* __restrict__ contrib)
{
    __shared__ __align__(16) __bf16 sA[2][MT * LDST];
    __shared__ __align__(16) __bf16 sB[2][NT * LDST];

    int z = blockIdx.z;
    int k = z & (Kc - 1);
    int b = z >> 3;
    int s0 = blockIdx.x * MT;
    int d0 = blockIdx.y * NT;

    const __bf16* Ag = hid + (size_t)z * Sc * Fc + (size_t)s0 * Fc;
    const __bf16* Bg = W2T + (size_t)k * Dc * Fc + (size_t)d0 * Fc;

    int tid = threadIdx.x, lane = tid & 31, wv = tid >> 5;
    int m0 = (wv & 1) * 32;
    int n0 = (wv >> 1) * 32;
    v8f c00 = {}, c01 = {}, c10 = {}, c11 = {};

    const int arow = tid >> 2, acb = (tid & 3) * 16;
    const int brow = tid >> 1, bcb = (tid & 1) * 32;

    GEMM_BODY(Fc, Fc)

    int col = lane & 15;
    int rbase = (lane & 16) ? 8 : 0;
    v8f* cs[4] = { &c00, &c01, &c10, &c11 };
#pragma unroll
    for (int t = 0; t < 4; ++t) {
        int mrow = s0 + m0 + (t >> 1) * 16 + rbase;     // s base
        int dcol = d0 + n0 + (t & 1) * 16 + col;
        float bias = b2[k * Dc + dcol];
#pragma unroll
        for (int i = 0; i < 8; ++i) {
            int row = mrow + i;
            float unm = masked[b * Sc + row] ? 0.f : 1.f;
            contrib[((size_t)(b * Sc + row) * Kc + k) * Dc + dcol] =
                unm * ((*cs[t])[i] + bias);
        }
    }
}

// ---------------------------------------------------------------------------
// Kernel 4: running ±R window sum (excluding self) over s, per (b,k,d), then
// mix over k with routing weights / max(cnt,1). Writes mixed -> d_out (f32).
// ---------------------------------------------------------------------------
constexpr int CH = 128;

__global__ __launch_bounds__(256) void mix_kernel(
    const float* __restrict__ contrib, const float* __restrict__ w,
    const float* __restrict__ cnt, float* __restrict__ out)
{
    int b = blockIdx.z;
    int d = blockIdx.x * 256 + threadIdx.x;
    int s0 = blockIdx.y * CH;

    float run[Kc];
#pragma unroll
    for (int e = 0; e < Kc; ++e) run[e] = 0.f;

    int lo = max(s0 - Rc, 0), hi = min(s0 + Rc, Sc - 1);
    for (int s2 = lo; s2 <= hi; ++s2) {
        const float* p = contrib + (size_t)(b * Sc + s2) * Kc * Dc + d;
#pragma unroll
        for (int e = 0; e < Kc; ++e) run[e] += p[(size_t)e * Dc];
    }

    for (int s = s0; s < s0 + CH; ++s) {
        if (s > s0) {
            if (s + Rc < Sc) {
                const float* p = contrib + (size_t)(b * Sc + s + Rc) * Kc * Dc + d;
#pragma unroll
                for (int e = 0; e < Kc; ++e) run[e] += p[(size_t)e * Dc];
            }
            if (s - Rc - 1 >= 0) {
                const float* p = contrib + (size_t)(b * Sc + s - Rc - 1) * Kc * Dc + d;
#pragma unroll
                for (int e = 0; e < Kc; ++e) run[e] -= p[(size_t)e * Dc];
            }
        }
        const float* pc = contrib + (size_t)(b * Sc + s) * Kc * Dc + d;
        const float* pw = w + (size_t)(b * Sc + s) * Kc;
        float invc = 1.f / fmaxf(cnt[b * Sc + s], 1.f);
        float m = 0.f;
#pragma unroll
        for (int e = 0; e < Kc; ++e) m += pw[e] * (run[e] - pc[(size_t)e * Dc]);
        out[(size_t)(b * Sc + s) * Dc + d] = m * invc;
    }
}

// ---------------------------------------------------------------------------
// Kernel 5: in-place LayerNorm (eps=1e-5, no affine) over D + validity mask.
// ---------------------------------------------------------------------------
__global__ __launch_bounds__(256) void ln_kernel(
    float* __restrict__ out, const unsigned char* __restrict__ masked,
    const float* __restrict__ cnt)
{
    __shared__ float s_sum[8], s_sq[8];
    int token = blockIdx.x;
    int tid = threadIdx.x, lane = tid & 31, wv = tid >> 5;
    float* p = out + (size_t)token * Dc;

    float v[8], sum = 0.f, sq = 0.f;
#pragma unroll
    for (int j = 0; j < 8; ++j) {
        float x = p[j * 256 + tid];
        v[j] = x; sum += x; sq += x * x;
    }
#pragma unroll
    for (int off = 16; off; off >>= 1) {
        sum += __shfl_xor(sum, off, 32);
        sq  += __shfl_xor(sq,  off, 32);
    }
    if (lane == 0) { s_sum[wv] = sum; s_sq[wv] = sq; }
    __syncthreads();
    if (tid == 0) {
        float a = 0.f, b2 = 0.f;
#pragma unroll
        for (int i = 0; i < 8; ++i) { a += s_sum[i]; b2 += s_sq[i]; }
        s_sum[0] = a; s_sq[0] = b2;
    }
    __syncthreads();
    float mu  = s_sum[0] * (1.f / Dc);
    float var = s_sq[0] * (1.f / Dc) - mu * mu;
    float rstd = rsqrtf(var + 1e-5f);
    bool valid = (masked[token] != 0) && (cnt[token] > 0.f);
#pragma unroll
    for (int j = 0; j < 8; ++j)
        p[j * 256 + tid] = valid ? (v[j] - mu) * rstd : 0.f;
}

// ---------------------------------------------------------------------------
// Launch. Inputs (setup_inputs order):
//   0 h_L[B,S,D] f32, 1 masked[B,S] bool(1B), 2 W_route[D,K] f32, 3 b_route[K],
//   4 W1[K,D,F] f32, 5 b1[K,F], 6 W2[K,F,D] f32, 7 b2[K,D], 8 range_r (unused, R=8)
// Workspace (~352 MB): w | cnt | hid | contrib | hA | W1T | W2T
// ---------------------------------------------------------------------------
extern "C" void kernel_launch(void* const* d_in, const int* in_sizes, int n_in,
                              void* d_out, int out_size, void* d_ws, size_t ws_size,
                              hipStream_t stream) {
    const float* h_L   = (const float*)d_in[0];
    const unsigned char* masked = (const unsigned char*)d_in[1];
    const float* W_route = (const float*)d_in[2];
    const float* b_route = (const float*)d_in[3];
    const float* W1 = (const float*)d_in[4];
    const float* b1 = (const float*)d_in[5];
    const float* W2 = (const float*)d_in[6];
    const float* b2 = (const float*)d_in[7];
    float* out = (float*)d_out;

    char* ws = (char*)d_ws;
    size_t off = 0;
    float*  w_rt = (float*)(ws + off);  off += (size_t)Bc * Sc * Kc * sizeof(float);
    float*  cnt  = (float*)(ws + off);  off += (size_t)Bc * Sc * sizeof(float);
    off = (off + 255) & ~(size_t)255;
    __bf16* hid  = (__bf16*)(ws + off); off += (size_t)Bc * Kc * Sc * Fc * sizeof(__bf16);
    off = (off + 255) & ~(size_t)255;
    float* contrib = (float*)(ws + off); off += (size_t)Bc * Sc * Kc * Dc * sizeof(float);
    off = (off + 255) & ~(size_t)255;
    __bf16* hA  = (__bf16*)(ws + off);  off += (size_t)Bc * Sc * Dc * sizeof(__bf16);
    off = (off + 255) & ~(size_t)255;
    __bf16* W1T = (__bf16*)(ws + off);  off += (size_t)Kc * Dc * Fc * sizeof(__bf16);
    off = (off + 255) & ~(size_t)255;
    __bf16* W2T = (__bf16*)(ws + off);  off += (size_t)Kc * Fc * Dc * sizeof(__bf16);
    (void)ws_size; (void)in_sizes; (void)n_in; (void)out_size;

    // Pre-passes: bf16 conversion (h_L) and transpose+convert (W1, W2).
    cvt_kernel<<<dim3((Bc * Sc * Dc) / (256 * 8)), dim3(256), 0, stream>>>(h_L, hA);
    // W1 [k][D][F] -> W1T [k][F][D]
    transpose_cvt_kernel<<<dim3(Fc / 64, Dc / 64, Kc), dim3(256), 0, stream>>>(W1, W1T, Dc, Fc);
    // W2 [k][F][D] -> W2T [k][D][F]
    transpose_cvt_kernel<<<dim3(Dc / 64, Fc / 64, Kc), dim3(256), 0, stream>>>(W2, W2T, Fc, Dc);

    route_kernel<<<dim3(Bc * Sc / 8), dim3(256), 0, stream>>>(
        h_L, masked, W_route, b_route, w_rt, cnt);

    gemm1_kernel<<<dim3(Sc / MT, Fc / NT, Bc * Kc), dim3(256), 0, stream>>>(
        hA, W1T, b1, hid);

    gemm2_kernel<<<dim3(Sc / MT, Dc / NT, Bc * Kc), dim3(256), 0, stream>>>(
        hid, W2T, b2, masked, contrib);

    mix_kernel<<<dim3(Dc / 256, Sc / CH, Bc), dim3(256), 0, stream>>>(
        contrib, w_rt, cnt, out);

    ln_kernel<<<dim3(Bc * Sc), dim3(256), 0, stream>>>(out, masked, cnt);
}
